// Global_Attention_12910671691947
// MI455X (gfx1250) — compile-verified, matrix operand order checked by reading, not run on hardware
//
#include <hip/hip_runtime.h>
#include <hip/hip_bf16.h>

typedef __bf16 bf16_t;
typedef bf16_t v16bf __attribute__((ext_vector_type(16)));
typedef bf16_t v8bf  __attribute__((ext_vector_type(8)));
typedef float  v8f   __attribute__((ext_vector_type(8)));
typedef float  v4f   __attribute__((ext_vector_type(4)));
typedef unsigned int v4u __attribute__((ext_vector_type(4)));
typedef int    v8i   __attribute__((ext_vector_type(8)));
typedef int    v4i   __attribute__((ext_vector_type(4)));

#define ET 12
#define ID 64
#define NB 8
#define SS 1024
#define HH 768
#define NN 1536   /* ET*ID*2 */
#define MM 8192   /* NB*SS   */

#if defined(__HIP_DEVICE_COMPILE__) && __has_builtin(__builtin_amdgcn_tensor_load_to_lds) && __has_builtin(__builtin_amdgcn_s_wait_tensorcnt)
#define USE_TDM 1
#else
#define USE_TDM 0
#endif

// Load one 16-lane-striped bf16 WMMA fragment: lane holds row (caller picks row),
// 8 elems at elemOff and 8 elems at elemOff+16 (per CDNA5 16-bit A/B VGPR layout).
__device__ __forceinline__ v16bf load_frag(const bf16_t* rowPtr, int elemOff) {
  v8bf lo = *(const v8bf*)(rowPtr + elemOff);
  v8bf hi = *(const v8bf*)(rowPtr + elemOff + 16);
  return __builtin_shufflevector(lo, hi, 0,1,2,3,4,5,6,7,8,9,10,11,12,13,14,15);
}

#if USE_TDM
// 2-D tensor_load_to_lds: tile (tile_d0 elems x tile_d1 rows), 2-byte elements,
// row stride stride0 elems; D# packed per CDNA5 ISA 8.3/8.4 bit layout.
__device__ __forceinline__ void tdm_load_2d(unsigned lds_off, const void* gptr,
                                            unsigned tensor_d0, unsigned tensor_d1,
                                            unsigned tile_d0, unsigned tile_d1,
                                            unsigned long long stride0) {
  unsigned long long ga = (unsigned long long)gptr;
  v4u g0;
  g0.x = 1u;                                   // count=1, user descriptor
  g0.y = lds_off;                              // lds_addr
  g0.z = (unsigned)(ga & 0xFFFFFFFFu);         // global_addr[31:0]
  g0.w = (unsigned)((ga >> 32) & 0x1FFFFFFu) | (2u << 30); // addr[56:32], type=2
  v8i g1;
  g1[0] = (int)(1u << 16);                     // wg_mask=0, data_size=1 (2B)
  g1[1] = (int)((tensor_d0 & 0xFFFFu) << 16);
  g1[2] = (int)((tensor_d0 >> 16) | ((tensor_d1 & 0xFFFFu) << 16));
  g1[3] = (int)((tensor_d1 >> 16) | (tile_d0 << 16));
  g1[4] = (int)(tile_d1 & 0xFFFFu);            // tile_dim1, tile_dim2=0
  g1[5] = (int)(stride0 & 0xFFFFFFFFu);
  g1[6] = (int)((stride0 >> 32) & 0xFFFFu);
  g1[7] = 0;
  v4i z4 = {0, 0, 0, 0};
  v8i z8 = {0, 0, 0, 0, 0, 0, 0, 0};
  __builtin_amdgcn_tensor_load_to_lds(g0, g1, z4, z4, z8, 0);
}
#endif

// ---------------- prep kernels ----------------
__global__ void k_cvt_hidden(const float* __restrict__ src, bf16_t* __restrict__ dst, int n4) {
  int i = blockIdx.x * blockDim.x + threadIdx.x;
  if (i < n4) {
    v4f v = *(const v4f*)(src + (size_t)i * 4);
    bf16_t* d = dst + (size_t)i * 4;
    d[0] = (bf16_t)v.x; d[1] = (bf16_t)v.y; d[2] = (bf16_t)v.z; d[3] = (bf16_t)v.w;
  }
}

__global__ void k_wt(const float* __restrict__ W, bf16_t* __restrict__ Wt) {
  int i = blockIdx.x * 256 + threadIdx.x;     // 0 .. NN*HH-1
  int n = i / HH, k = i - n * HH;
  Wt[i] = (bf16_t)W[(size_t)k * NN + n];
}

__global__ void k_tab(float* __restrict__ st, float* __restrict__ ct) {
  int i = blockIdx.x * 256 + threadIdx.x;     // 0 .. SS*32-1
  int pos = i >> 5, fi = i & 31;
  float invf = expf(-0.28782313662425572f * (float)fi); // 10000^(-2*fi/64)
  float ang = (float)pos * invf;
  st[i] = sinf(ang);
  ct[i] = cosf(ang);
}

// ---------------- GEMM1 + bias + RoPE -> Qb/Kb (bf16) ----------------
// WG: 256 thr (8 waves), tile 128(M) x 128(N), K-step 32.
// wave(wm,wn): wm=w&1 -> 64 M rows, wn=w>>1 -> 32 N cols => 8 C tiles/wave.
__global__ void __launch_bounds__(256) k_proj_rope(
    const bf16_t* __restrict__ Hb, const bf16_t* __restrict__ Wt,
    const float* __restrict__ bias,
    const float* __restrict__ sinT, const float* __restrict__ cosT,
    bf16_t* __restrict__ Qb, bf16_t* __restrict__ Kb)
{
  __shared__ __align__(16) bf16_t Als[128 * 32];
  __shared__ __align__(16) bf16_t Bls[128 * 32];
  const int tid = threadIdx.x;
  const int w = tid >> 5, l = tid & 31, ln = l & 15, hl = l >> 4;
  const int mBase = blockIdx.y * 128;
  const int nBase = blockIdx.x * 128;
  const int m0w = (w & 1) * 64, n0w = (w >> 1) * 32;

  v8f acc[4][2] = {};

  for (int kk = 0; kk < 24; ++kk) {
    const int k0 = kk * 32;
    __syncthreads();                       // LDS free to overwrite
#if USE_TDM
    if (tid < 32) {
      tdm_load_2d((unsigned)(unsigned long long)(void*)Als,
                  Hb + (size_t)mBase * HH + k0, HH, MM, 32, 128, HH);
      tdm_load_2d((unsigned)(unsigned long long)(void*)Bls,
                  Wt + (size_t)nBase * HH + k0, HH, NN, 32, 128, HH);
      __builtin_amdgcn_s_wait_tensorcnt(0);
    }
#else
    for (int c = tid; c < 512; c += 256) { // 128 rows x 4 16B-chunks
      int row = c >> 2, q = c & 3;
      *(v8bf*)(Als + row * 32 + q * 8) =
          *(const v8bf*)(Hb + (size_t)(mBase + row) * HH + k0 + q * 8);
      *(v8bf*)(Bls + row * 32 + q * 8) =
          *(const v8bf*)(Wt + (size_t)(nBase + row) * HH + k0 + q * 8);
    }
#endif
    __syncthreads();                       // tile ready

    v16bf af[4], bfr[2];
#pragma unroll
    for (int mi = 0; mi < 4; ++mi)
      af[mi] = load_frag(Als + (m0w + 16 * mi + ln) * 32, 8 * hl);
#pragma unroll
    for (int nj = 0; nj < 2; ++nj)
      bfr[nj] = load_frag(Bls + (n0w + 16 * nj + ln) * 32, 8 * hl);
#pragma unroll
    for (int mi = 0; mi < 4; ++mi)
#pragma unroll
      for (int nj = 0; nj < 2; ++nj)
        acc[mi][nj] = __builtin_amdgcn_wmma_f32_16x16x32_bf16(
            false, af[mi], false, bfr[nj], (short)0, acc[mi][nj], false, false);
  }

  // epilogue: bias + interleaved RoPE (pairs = adjacent lanes), cvt bf16, store
#pragma unroll
  for (int nj = 0; nj < 2; ++nj) {
    const int ng = nBase + n0w + 16 * nj + ln;   // global column
    const int hIdx = ng >> 7, r = ng & 127;
    const bool isK = r >= 64;
    const int d = r & 63, fi = d >> 1;
    const float bv = bias[ng];
    bf16_t* dstBase = isK ? Kb : Qb;
#pragma unroll
    for (int mi = 0; mi < 4; ++mi)
#pragma unroll
      for (int v = 0; v < 8; ++v) {
        const int mg = mBase + m0w + 16 * mi + v + 8 * hl;
        const int pos = mg & (SS - 1), bb = mg >> 10;
        float x = acc[mi][nj][v] + bv;
        float p = __shfl_xor(x, 1, 32);          // pair element (col n^1)
        float cc = cosT[pos * 32 + fi], ss = sinT[pos * 32 + fi];
        float y = (ng & 1) ? (x * cc + p * ss) : (x * cc - p * ss);
        dstBase[(((size_t)(bb * ET + hIdx) * SS + pos) << 6) + d] = (bf16_t)y;
      }
  }
}

// ---------------- batched Q·K^T, coalesced bmnh store ----------------
// WG: 384 thr = 12 waves, wave w = head h. Wave tile: 16(M) x 64(N), K=64.
__global__ void __launch_bounds__(384) k_attn(
    const bf16_t* __restrict__ Qb, const bf16_t* __restrict__ Kb,
    float* __restrict__ out)
{
  __shared__ __align__(16) float Ols[16 * 768];  // [m'][n'*12+h], 48 KB
  const int tid = threadIdx.x;
  const int w = tid >> 5, l = tid & 31, ln = l & 15, hl = l >> 4;
  const int bb = blockIdx.y;
  const int mt = blockIdx.x >> 4;      // 0..63 (M tile of 16)
  const int ng = blockIdx.x & 15;      // 0..15 (N group of 64)
  const size_t hbase = (size_t)(bb * ET + w) * SS * ID;

  const bf16_t* qrow = Qb + hbase + (size_t)(mt * 16 + ln) * ID;
  v16bf alo = load_frag(qrow, 8 * hl);
  v16bf ahi = load_frag(qrow, 32 + 8 * hl);

#pragma unroll
  for (int j = 0; j < 4; ++j) {
    const bf16_t* krow = Kb + hbase + (size_t)(ng * 64 + j * 16 + ln) * ID;
    v16bf blo = load_frag(krow, 8 * hl);
    v16bf bhi = load_frag(krow, 32 + 8 * hl);
    v8f acc = {};
    acc = __builtin_amdgcn_wmma_f32_16x16x32_bf16(false, alo, false, blo,
                                                  (short)0, acc, false, false);
    acc = __builtin_amdgcn_wmma_f32_16x16x32_bf16(false, ahi, false, bhi,
                                                  (short)0, acc, false, false);
#pragma unroll
    for (int v = 0; v < 8; ++v)
      Ols[(v + 8 * hl) * 768 + (j * 16 + ln) * ET + w] = acc[v] * 0.125f;
  }
  __syncthreads();

  // coalesced store: tile = 16 rows x 768 floats (row-contiguous in bmnh)
  const size_t obase = (((size_t)(bb * SS + mt * 16)) * SS + (size_t)ng * 64) * ET;
#pragma unroll
  for (int i = 0; i < 8; ++i) {
    int f = (i * 384 + tid) * 4;
    int mrow = f / 768;
    int rr = f - mrow * 768;
    *(v4f*)(out + obase + (size_t)mrow * (SS * ET) + rr) = *(const v4f*)(&Ols[f]);
  }
}

extern "C" void kernel_launch(void* const* d_in, const int* in_sizes, int n_in,
                              void* d_out, int out_size, void* d_ws, size_t ws_size,
                              hipStream_t stream) {
  (void)in_sizes; (void)n_in; (void)out_size; (void)ws_size;
  const float* hidden = (const float*)d_in[0];
  const float* W      = (const float*)d_in[1];
  const float* bias   = (const float*)d_in[2];
  float* out = (float*)d_out;

  char* ws = (char*)d_ws;
  size_t off = 0;
  bf16_t* Hb = (bf16_t*)(ws + off); off += (size_t)MM * HH * 2;          // 12.6 MB
  bf16_t* Wt = (bf16_t*)(ws + off); off += (size_t)NN * HH * 2;          //  2.4 MB
  bf16_t* Qb = (bf16_t*)(ws + off); off += (size_t)NB * ET * SS * ID * 2; // 12.6 MB
  bf16_t* Kb = (bf16_t*)(ws + off); off += (size_t)NB * ET * SS * ID * 2; // 12.6 MB
  float*  st = (float*)(ws + off);  off += (size_t)SS * 32 * 4;
  float*  ct = (float*)(ws + off);  off += (size_t)SS * 32 * 4;           // ~40.4 MB total

  k_cvt_hidden<<<(MM * HH / 4 + 255) / 256, 256, 0, stream>>>(hidden, Hb, MM * HH / 4);
  k_wt<<<(NN * HH) / 256, 256, 0, stream>>>(W, Wt);
  k_tab<<<(SS * 32) / 256, 256, 0, stream>>>(st, ct);
  k_proj_rope<<<dim3(NN / 128, MM / 128), 256, 0, stream>>>(Hb, Wt, bias, st, ct, Qb, Kb);
  k_attn<<<dim3(64 * 16, NB), 384, 0, stream>>>(Qb, Kb, out);
}
